// RNN_2_6966436954271
// MI455X (gfx1250) — compile-verified, matrix operand order checked by reading, not run on hardware
//
#include <hip/hip_runtime.h>
#include <hip/hip_bf16.h>
#include <stdint.h>

// ---------------------------------------------------------------------------
// LSTM (T=65536, IN=320, H=256) + MLP head, CDNA5 / gfx1250 (wave32, WMMA)
//  - x_proj GEMM: bf16 WMMA, memory bound (~352 MB @ 23.3 TB/s ~= 15us)
//  - scan: W_hh resident in VGPRs (16 WMMA frags/wave * 32 waves = 512KB bf16),
//    16 v_wmma per wave per step; x_proj rows streamed one step ahead by the
//    Tensor Data Mover into an LDS double buffer (TENSORcnt synchronized).
// ---------------------------------------------------------------------------

typedef __attribute__((ext_vector_type(16))) __bf16 v16bf;
typedef __attribute__((ext_vector_type(8)))  __bf16 v8bf;
typedef __attribute__((ext_vector_type(8)))  float  v8f;
typedef __attribute__((ext_vector_type(4)))  unsigned int v4u;
typedef __attribute__((ext_vector_type(8)))  int    v8i;
typedef __attribute__((ext_vector_type(4)))  int    v4i;

#define T_STEPS 65536
#define IN_DIM  320
#define H_DIM   256
#define G_DIM   1024   // 4*H
#define H1_DIM  128
#define OUT_DIM 63

// workspace layout (bytes)
#define XP_OFF    0ull                                          // T*G f32 = 256 MB
#define WIHB_OFF  (XP_OFF   + (size_t)T_STEPS * G_DIM * 4)      // G*IN bf16
#define WHHB_OFF  (WIHB_OFF + (size_t)G_DIM * IN_DIM * 2)       // G*H  bf16
#define BSUM_OFF  (WHHB_OFF + (size_t)G_DIM * H_DIM * 2)        // G    f32
#define HLAST_OFF (BSUM_OFF + (size_t)G_DIM * 4)                // H    f32

static __device__ __forceinline__ uint16_t bfbits(float f) {
    union { float f; uint32_t u; } v; v.f = f;
    uint32_t r = (v.u + 0x7FFFu + ((v.u >> 16) & 1u)) >> 16;   // RNE
    return (uint16_t)r;
}

static __device__ __forceinline__ v16bf cat16(v8bf lo, v8bf hi) {
    return __builtin_shufflevector(lo, hi, 0,1,2,3,4,5,6,7,8,9,10,11,12,13,14,15);
}

static __device__ __forceinline__ v8f wmma_bf16(v16bf a, v16bf b, v8f c) {
    // D = A(16x32 bf16) * B(32x16 bf16) + C(16x16 f32)
    return __builtin_amdgcn_wmma_f32_16x16x32_bf16(false, a, false, b,
                                                   (short)0, c, false, false);
}

static __device__ __forceinline__ float sigm(float x) {
    return 1.0f / (1.0f + __expf(-x));
}

// ---------------------------------------------------------------------------
// Tensor Data Mover: async copy of one x_proj row (1024 f32 = 4KB, contiguous)
// into LDS. D# per CDNA5 ISA ch.8: 2D tile {tile_dim0=1024, tile_dim1=1}.
// ---------------------------------------------------------------------------
#if __has_builtin(__builtin_amdgcn_tensor_load_to_lds)
#define HAVE_TDM 1
static __device__ __forceinline__ void tdm_load_row(const float* gsrc,
                                                    uint32_t lds_off) {
    const uint64_t ga = (uint64_t)(uintptr_t)gsrc;
    v4u g0;
    g0[0] = 1u;                                            // count=1 (valid D#)
    g0[1] = lds_off;                                       // lds_addr
    g0[2] = (uint32_t)ga;                                  // global_addr[31:0]
    g0[3] = (uint32_t)((ga >> 32) & 0x01FFFFFFu)           // global_addr[56:32]
          | (2u << 30);                                    // type = 2 ("image")
    v8i g1;
    g1[0] = (int)(2u << 16);                               // data_size = 4B
    g1[1] = (int)(((uint32_t)G_DIM & 0xFFFFu) << 16);      // tensor_dim0 = 1024
    g1[2] = (int)(1u << 16);                               // tensor_dim1 = 1
    g1[3] = (int)((uint32_t)G_DIM << 16);                  // tile_dim0 = 1024
    g1[4] = 1;                                             // tile_dim1=1, tile_dim2=0
    g1[5] = G_DIM;                                         // tensor_dim0_stride
    g1[6] = 0;                                             // tensor_dim1_stride lo
    g1[7] = 0;
    v4i g2; g2[0] = g2[1] = g2[2] = g2[3] = 0;             // 2D: groups 2/3 unused
    v4i g3; g3[0] = g3[1] = g3[2] = g3[3] = 0;
#if __clang_major__ >= 23
    v8i g4; g4[0]=g4[1]=g4[2]=g4[3]=g4[4]=g4[5]=g4[6]=g4[7]=0;
    __builtin_amdgcn_tensor_load_to_lds(g0, g1, g2, g3, g4, 0);
#else
    __builtin_amdgcn_tensor_load_to_lds(g0, g1, g2, g3, 0);
#endif
}
#else
#define HAVE_TDM 0
#endif

// ---------------------------------------------------------------------------
// Kernel 0: weights -> bf16, bias fold (b_ih + b_hh)
// ---------------------------------------------------------------------------
__global__ void convert_kernel(const float* __restrict__ wih,
                               const float* __restrict__ whh,
                               const float* __restrict__ bih,
                               const float* __restrict__ bhh,
                               uint16_t* __restrict__ wihb,
                               uint16_t* __restrict__ whhb,
                               float* __restrict__ bsum) {
    int idx = blockIdx.x * blockDim.x + threadIdx.x;
    const int N0 = G_DIM * IN_DIM;
    const int N1 = N0 + G_DIM * H_DIM;
    const int N2 = N1 + G_DIM;
    if (idx < N0) {
        wihb[idx] = bfbits(wih[idx]);
    } else if (idx < N1) {
        int k = idx - N0;
        whhb[k] = bfbits(whh[k]);
    } else if (idx < N2) {
        int k = idx - N1;
        bsum[k] = bih[k] + bhh[k];
    }
}

// ---------------------------------------------------------------------------
// Kernel 1: x_proj[T,1024] = inputs[T,320] @ W_ih^T + bsum   (bf16 WMMA)
// One wave -> 16x64 output strip (4 accumulators, A fragment reused 4x).
// ---------------------------------------------------------------------------
__global__ void __launch_bounds__(256) xproj_gemm(const float*    __restrict__ x,
                                                  const uint16_t* __restrict__ wihb,
                                                  const float*    __restrict__ bsum,
                                                  float*          __restrict__ xp) {
    const int lane = threadIdx.x & 31;
    const int wave = blockIdx.x * (blockDim.x >> 5) + (threadIdx.x >> 5);
    const int rt = wave >> 4;          // 0..4095 row tile (16 rows)
    const int cs = wave & 15;          // 0..15   col strip (64 cols)
    const int row0 = rt * 16;
    const int col0 = cs * 64;

    const int m    = lane & 15;            // row / col within tile
    const int s    = (lane >> 4) * 8;      // K sub-offset per lane half

    v8f acc[4];
    #pragma unroll
    for (int n = 0; n < 4; ++n)
        #pragma unroll
        for (int i = 0; i < 8; ++i) acc[n][i] = 0.0f;

    for (int k0 = 0; k0 < IN_DIM; k0 += 32) {
        // A fragment: 16x32 bf16 from fp32 inputs (lane m = row, K {s..s+7, s+16..s+23})
        const float* ap = x + (size_t)(row0 + m) * IN_DIM + k0 + s;
        v8f alo = *(const v8f*)ap;
        v8f ahi = *(const v8f*)(ap + 16);
        union { v16bf v; uint16_t u[16]; } a;
        #pragma unroll
        for (int i = 0; i < 8; ++i) {
            a.u[i]     = bfbits(alo[i]);
            a.u[i + 8] = bfbits(ahi[i]);
        }
        #pragma unroll
        for (int n = 0; n < 4; ++n) {
            // B fragment: column j = W_ih row j (contiguous in K)
            const uint16_t* bp =
                wihb + (size_t)(col0 + n * 16 + m) * IN_DIM + k0 + s;
            v8bf blo = *(const v8bf*)bp;
            v8bf bhi = *(const v8bf*)(bp + 16);
            acc[n] = wmma_bf16(a.v, cat16(blo, bhi), acc[n]);
        }
    }

    // D layout: vgpr v -> row v (lanes 0-15) / row v+8 (lanes 16-31), col = lane&15
    #pragma unroll
    for (int n = 0; n < 4; ++n) {
        const int c = col0 + n * 16 + m;
        const float b = bsum[c];
        #pragma unroll
        for (int v = 0; v < 8; ++v) {
            const int r = row0 + v + ((lane >> 4) << 3);
            xp[(size_t)r * G_DIM + c] = acc[n][v] + b;
        }
    }
}

// ---------------------------------------------------------------------------
// Kernel 2: sequential LSTM scan. Single workgroup, 1024 threads = 32 waves.
// Wave w holds gate rows [32w, 32w+32) of W_hh as 16 resident WMMA A-fragments
// (128 VGPRs/lane => whole 512KB bf16 W_hh lives in the register file).
// Per step: TDM prefetches x_proj[t+1] into LDS; B = h broadcast into all 16
// columns; 16 WMMAs/wave; contiguous v8f gate rows stored with ds_store_b128.
// ---------------------------------------------------------------------------
__global__ void __launch_bounds__(1024) lstm_scan(const uint16_t* __restrict__ whhb,
                                                  const float*    __restrict__ xp,
                                                  float*          __restrict__ hlast) {
    __shared__ __align__(32) float    gbuf[G_DIM];
    __shared__ float                  cbuf[H_DIM];
    __shared__ __align__(16) uint16_t hbf[H_DIM];
#if HAVE_TDM
    __shared__ __align__(16) float    xbuf[2][G_DIM];   // TDM double buffer
#endif

    const int tid  = threadIdx.x;
    const int lane = tid & 31;
    const int w    = tid >> 5;            // wave 0..31
    const int m    = lane & 15;
    const int s    = (lane >> 4) * 8;

    // preload W_hh fragments (row tiles 2w, 2w+1; K tiles 0..7)
    v16bf A[2][8];
    #pragma unroll
    for (int half = 0; half < 2; ++half) {
        #pragma unroll
        for (int kt = 0; kt < 8; ++kt) {
            const int row = 32 * w + 16 * half + m;
            const uint16_t* p = whhb + (size_t)row * H_DIM + kt * 32 + s;
            v8bf lo = *(const v8bf*)p;
            v8bf hi = *(const v8bf*)(p + 16);
            A[half][kt] = cat16(lo, hi);
        }
    }

    if (tid < H_DIM) { cbuf[tid] = 0.0f; hbf[tid] = 0; }
#if HAVE_TDM
    if (w == 0)                                   // prime the pipeline: row 0
        tdm_load_row(xp, (uint32_t)(uintptr_t)&xbuf[0][0]);
#endif
    __syncthreads();

    for (int t = 0; t < T_STEPS; ++t) {
#if HAVE_TDM
        if (w == 0 && t + 1 < T_STEPS)            // async prefetch of row t+1
            tdm_load_row(xp + (size_t)(t + 1) * G_DIM,
                         (uint32_t)(uintptr_t)&xbuf[(t + 1) & 1][0]);
#endif
        // ---- gates = W_hh . h via WMMA (h replicated over all 16 B columns)
        v8f acc0, acc1;
        #pragma unroll
        for (int i = 0; i < 8; ++i) { acc0[i] = 0.0f; acc1[i] = 0.0f; }
        #pragma unroll
        for (int kt = 0; kt < 8; ++kt) {
            const uint16_t* hp = hbf + kt * 32 + s;     // ds_load, broadcast
            v8bf lo = *(const v8bf*)hp;
            v8bf hi = *(const v8bf*)(hp + 16);
            v16bf b = cat16(lo, hi);
            acc0 = wmma_bf16(A[0][kt], b, acc0);
            acc1 = wmma_bf16(A[1][kt], b, acc1);
        }
        if ((lane & 15) == 0) {                   // lanes 0 and 16: v8f = 8 rows
            const int rbase = 32 * w + ((lane >> 4) << 3);
            *(v8f*)(gbuf + rbase)      = acc0;    // ds_store_b128 x2
            *(v8f*)(gbuf + rbase + 16) = acc1;
        }
#if HAVE_TDM
        if (w == 0) {                             // row t complete (in-order TDM)
            if (t + 1 < T_STEPS) __builtin_amdgcn_s_wait_tensorcnt(1);
            else                 __builtin_amdgcn_s_wait_tensorcnt(0);
        }
#endif
        __syncthreads();

        // ---- cell update (torch gate order i,f,g,o), bias already in x_proj
#if HAVE_TDM
        const float* xpt = xbuf[t & 1];
#else
        const float* xpt = xp + (size_t)t * G_DIM;
#endif
        if (tid < H_DIM) {
            const float ig = sigm (gbuf[tid            ] + xpt[tid            ]);
            const float fg = sigm (gbuf[tid +     H_DIM] + xpt[tid +     H_DIM]);
            const float gg = tanhf(gbuf[tid + 2 * H_DIM] + xpt[tid + 2 * H_DIM]);
            const float og = sigm (gbuf[tid + 3 * H_DIM] + xpt[tid + 3 * H_DIM]);
            const float c  = fg * cbuf[tid] + ig * gg;
            cbuf[tid] = c;
            const float h = og * tanhf(c);
            hbf[tid] = bfbits(h);
            if (t == T_STEPS - 1) hlast[tid] = h;
        }
#if !HAVE_TDM
        else if (tid < H_DIM + 32) {
            if (t + 1 < T_STEPS)
                __builtin_prefetch(xp + (size_t)(t + 1) * G_DIM + (tid - H_DIM) * 32, 0, 1);
        }
#endif
        __syncthreads();
    }
}

// ---------------------------------------------------------------------------
// Kernel 3: head  out = relu(h W1^T + b1) W2^T + b2   (tiny, plain VALU)
// ---------------------------------------------------------------------------
__global__ void __launch_bounds__(128) head_kernel(const float* __restrict__ hlast,
                                                   const float* __restrict__ W1,
                                                   const float* __restrict__ b1,
                                                   const float* __restrict__ W2,
                                                   const float* __restrict__ b2,
                                                   float* __restrict__ out) {
    __shared__ float hl[H_DIM];
    __shared__ float r1[H1_DIM];
    const int tid = threadIdx.x;
    hl[tid]       = hlast[tid];
    hl[tid + 128] = hlast[tid + 128];
    __syncthreads();
    float acc = b1[tid];
    for (int k = 0; k < H_DIM; ++k) acc += W1[tid * H_DIM + k] * hl[k];
    r1[tid] = fmaxf(acc, 0.0f);
    __syncthreads();
    if (tid < OUT_DIM) {
        float a2 = b2[tid];
        for (int k = 0; k < H1_DIM; ++k) a2 += W2[tid * H1_DIM + k] * r1[k];
        out[tid] = a2;
    }
}

// ---------------------------------------------------------------------------
extern "C" void kernel_launch(void* const* d_in, const int* in_sizes, int n_in,
                              void* d_out, int out_size, void* d_ws, size_t ws_size,
                              hipStream_t stream) {
    const float* x   = (const float*)d_in[0];
    const float* Wih = (const float*)d_in[1];
    const float* Whh = (const float*)d_in[2];
    const float* bih = (const float*)d_in[3];
    const float* bhh = (const float*)d_in[4];
    const float* W1  = (const float*)d_in[5];
    const float* b1  = (const float*)d_in[6];
    const float* W2  = (const float*)d_in[7];
    const float* b2  = (const float*)d_in[8];

    char* ws = (char*)d_ws;
    float*    xp    = (float*)   (ws + XP_OFF);
    uint16_t* wihb  = (uint16_t*)(ws + WIHB_OFF);
    uint16_t* whhb  = (uint16_t*)(ws + WHHB_OFF);
    float*    bsum  = (float*)   (ws + BSUM_OFF);
    float*    hlast = (float*)   (ws + HLAST_OFF);

    const int convN = G_DIM * IN_DIM + G_DIM * H_DIM + G_DIM;
    convert_kernel<<<(convN + 255) / 256, 256, 0, stream>>>(Wih, Whh, bih, bhh,
                                                            wihb, whhb, bsum);

    // 4096 row tiles x 16 col strips = 65536 waves, 8 waves/block
    xproj_gemm<<<8192, 256, 0, stream>>>(x, wihb, bsum, xp);

    lstm_scan<<<1, 1024, 0, stream>>>(whhb, xp, hlast);

    head_kernel<<<1, 128, 0, stream>>>(hlast, W1, b1, W2, b2, (float*)d_out);
}
